// AttentionRNN_52974126629678
// MI455X (gfx1250) — matrix operand order repaired and checked by reading, not compiled, not run on hardware
//
#include <hip/hip_runtime.h>
#include <hip/hip_bf16.h>
#include <math.h>

// ---------------------------------------------------------------------------
// Problem constants
// ---------------------------------------------------------------------------
#define BB   64
#define TT   512
#define HH   512
#define MM   80
#define LOCD 32
#define LOCK 31
#define LOCP 15

// packed-weight chunk per N-chunk: 512 (loc) + 16*512 (annot_w) bf16 elements
#define WCHUNK_ELEMS 8704
#define WCHUNK_BYTES (WCHUNK_ELEMS * 2)   // 17408

typedef __attribute__((ext_vector_type(16))) __bf16 v16bf;
typedef __attribute__((ext_vector_type(8)))  __bf16 v8bf;
typedef __attribute__((ext_vector_type(8)))  float  v8f;
typedef __attribute__((ext_vector_type(4)))  unsigned int u32x4;
typedef __attribute__((ext_vector_type(8)))  int  i32x8;
typedef __attribute__((ext_vector_type(4)))  int  i32x4;

#if defined(__has_include)
#if __has_include(<hip/amd_detail/amd_gfx1250_TDM.h>)
#define TDM_6ARG 1
#endif
#endif

#if defined(__has_builtin)
#if __has_builtin(__builtin_amdgcn_tensor_load_to_lds)
#define HAVE_TDM 1
#endif
#endif

// ---------------------------------------------------------------------------
// Helpers
// ---------------------------------------------------------------------------
__device__ __forceinline__ v16bf cvt16(const float4& a, const float4& b,
                                       const float4& c, const float4& d) {
    v16bf t;
    t[0]  = (__bf16)a.x; t[1]  = (__bf16)a.y; t[2]  = (__bf16)a.z; t[3]  = (__bf16)a.w;
    t[4]  = (__bf16)b.x; t[5]  = (__bf16)b.y; t[6]  = (__bf16)b.z; t[7]  = (__bf16)b.w;
    t[8]  = (__bf16)c.x; t[9]  = (__bf16)c.y; t[10] = (__bf16)c.z; t[11] = (__bf16)c.w;
    t[12] = (__bf16)d.x; t[13] = (__bf16)d.y; t[14] = (__bf16)d.z; t[15] = (__bf16)d.w;
    return t;
}

#ifdef HAVE_TDM
// Issue a TDM DMA of one contiguous WCHUNK_BYTES weight chunk into LDS.
// D# built per CDNA5 ISA: 1 row, 8704 2-byte elements.
__device__ __forceinline__ void tdm_load_chunk(unsigned int lds_addr,
                                               unsigned long long gaddr) {
    u32x4 g0;
    g0[0] = 1u;                                   // count=1, no gather
    g0[1] = lds_addr;                             // lds_addr (bytes)
    g0[2] = (unsigned int)gaddr;                  // global_addr[31:0]
    g0[3] = (unsigned int)((gaddr >> 32) & 0x01FFFFFFu) | 0x80000000u; // [56:32] | type=2

    i32x8 g1;
    g1[0] = 0x00010000;                           // data_size=1 (2 bytes)
    g1[1] = (int)((WCHUNK_ELEMS & 0xFFFF) << 16); // tensor_dim0[15:0] in [31:16]
    g1[2] = (int)0x00010000;                      // tensor_dim0 hi=0 | tensor_dim1=1
    g1[3] = (int)((WCHUNK_ELEMS & 0xFFFF) << 16); // tile_dim0 in [31:16]
    g1[4] = 1;                                    // tile_dim1=1, tile_dim2=0
    g1[5] = WCHUNK_ELEMS;                         // tensor_dim0_stride lo
    g1[6] = 0;
    g1[7] = 0;

    i32x4 g2 = {0, 0, 0, 0};
    i32x4 g3 = {0, 0, 0, 0};
#ifdef TDM_6ARG
    i32x8 g4 = {0, 0, 0, 0, 0, 0, 0, 0};
    __builtin_amdgcn_tensor_load_to_lds(g0, g1, g2, g3, g4, 0);
#else
    __builtin_amdgcn_tensor_load_to_lds(g0, g1, g2, g3, 0);
#endif
}
#endif

// ---------------------------------------------------------------------------
// Kernel 1: pack loc_lin_w + annot_w into one combined bf16 buffer, laid out
// per N-chunk in this code's B-operand order (contiguous k, matching the A loads):
//   chunk nc (WCHUNK_ELEMS): [ locB (512) | kc=0..15 annotB (16*512) ]
//   element (l, j) of a sub-chunk = lane l's j-th bf16 (n = nc*16 + l%16,
//   k = (l/16)*16 + j within the sub-chunk's K range).
// ---------------------------------------------------------------------------
__global__ void pack_w_kernel(const float* __restrict__ aw,
                              const float* __restrict__ lw,
                              __bf16* __restrict__ out) {
    int idx = blockIdx.x * 256 + threadIdx.x;   // < 32 * WCHUNK_ELEMS
    int nc = idx / WCHUNK_ELEMS;
    int r  = idx - nc * WCHUNK_ELEMS;
    float v;
    if (r < 512) {                               // loc_lin_w chunk (K=32)
        int j = r & 15, l = (r >> 4) & 31;
        v = lw[(nc * 16 + (l & 15)) * LOCD + (l >> 4) * 16 + j];
    } else {                                     // annot_w chunks (K=512)
        int rb = r - 512;
        int kc = rb >> 9;
        int e  = rb & 511;
        int j = e & 15, l = e >> 4;
        v = aw[(size_t)(nc * 16 + (l & 15)) * HH + kc * 32 + (l >> 4) * 16 + j];
    }
    out[idx] = (__bf16)v;
}

// ---------------------------------------------------------------------------
// Kernel 2: location conv -> loc (B,T,32) bf16 (A-operand friendly row-major)
// ---------------------------------------------------------------------------
__global__ void loc_conv_kernel(const float* __restrict__ av,
                                const float* __restrict__ w,
                                __bf16* __restrict__ loc) {
    int idx = blockIdx.x * 256 + threadIdx.x;   // < B*T*32
    int c = idx & 31;
    int t = (idx >> 5) & (TT - 1);
    int b = idx >> 14;
    const float* w0 = w  + c * (2 * LOCK);
    const float* a0 = av + (size_t)b * 2 * TT;
    float s = 0.f;
    #pragma unroll
    for (int k = 0; k < LOCK; ++k) {
        int p = t + k - LOCP;
        if (p >= 0 && p < TT)
            s += a0[p] * w0[k] + a0[TT + p] * w0[LOCK + k];
    }
    loc[idx] = (__bf16)s;
}

// ---------------------------------------------------------------------------
// Kernel 3: fused GRUCell (torch semantics).  One block per batch row.
// ---------------------------------------------------------------------------
__global__ void gru_kernel(const float* __restrict__ mem,
                           const float* __restrict__ ctx,
                           const float* __restrict__ hstate,
                           const float* __restrict__ Wih,
                           const float* __restrict__ Whh,
                           const float* __restrict__ bih,
                           const float* __restrict__ bhh,
                           float* __restrict__ out) {
    __shared__ float xs[MM + HH];
    __shared__ float hs[HH];
    int b = blockIdx.x;
    int t = threadIdx.x;            // 512
    if (t < MM) xs[t] = mem[b * MM + t];
    xs[MM + t] = ctx[b * HH + t];
    hs[t]      = hstate[b * HH + t];
    __syncthreads();

    float gi[3], gh[3];
    #pragma unroll
    for (int g = 0; g < 3; ++g) {
        const float* wr = Wih + (size_t)(g * HH + t) * (MM + HH);
        float s = bih[g * HH + t];
        for (int k = 0; k < MM + HH; ++k) s += xs[k] * wr[k];
        gi[g] = s;
        const float* wh = Whh + (size_t)(g * HH + t) * HH;
        float s2 = bhh[g * HH + t];
        for (int k = 0; k < HH; ++k) s2 += hs[k] * wh[k];
        gh[g] = s2;
    }
    float r = 1.f / (1.f + expf(-(gi[0] + gh[0])));
    float z = 1.f / (1.f + expf(-(gi[1] + gh[1])));
    float n = tanhf(gi[2] + r * gh[2]);
    out[b * HH + t] = (1.f - z) * n + z * hs[t];
}

// ---------------------------------------------------------------------------
// Kernel 4: qcomb[b,h] = rnn_out[b,:]·query_w[h,:] + query_b + annot_b + loc_b
// ---------------------------------------------------------------------------
__global__ void qcomb_kernel(const float* __restrict__ rnn_out,
                             const float* __restrict__ qw,
                             const float* __restrict__ qb,
                             const float* __restrict__ ab,
                             const float* __restrict__ lb,
                             float* __restrict__ qcomb) {
    __shared__ float rs[HH];
    int b = blockIdx.x, t = threadIdx.x;        // 512
    rs[t] = rnn_out[b * HH + t];
    __syncthreads();
    const float* w = qw + (size_t)t * HH;
    float s = qb[t] + ab[t] + lb[t];
    for (int k = 0; k < HH; ++k) s += rs[k] * w[k];
    qcomb[b * HH + t] = s;
}

// ---------------------------------------------------------------------------
// Kernel 5: fused attention-energy GEMM (17.2 GFLOP hot spot).
//  - 8 waves/block, one 16-row (b,t) tile per wave; A tile (16x512 bf16)
//    resident in VGPRs (launch_bounds prevents spilling).
//  - shared B operands staged to LDS by the Tensor Data Mover, double
//    buffered: TDM for chunk nc+1 overlaps the 17 WMMAs of chunk nc.
// ---------------------------------------------------------------------------
__global__ __launch_bounds__(256, 1)
void energy_kernel(const float*  __restrict__ annot,   // (B*T, H) f32
                   const __bf16* __restrict__ locA,    // (B*T, 32) bf16
                   const __bf16* __restrict__ packW,   // combined packed weights
                   const float*  __restrict__ qcomb,   // (B, H)
                   const float*  __restrict__ vw,      // (H)
                   float* __restrict__ energy) {       // (B*T)
    __shared__ __align__(16) unsigned char stage[2][WCHUNK_BYTES];

    const int lane  = threadIdx.x & 31;
    const int wave  = threadIdx.x >> 5;
    const int tile  = blockIdx.x * 8 + wave;   // < 2048
    const int m     = lane & 15;
    const int khalf = lane >> 4;
    const int bt0   = tile * 16;
    const int b     = bt0 >> 9;                // / T

    // --- A tile for annotations: 16 chunks of K=32, resident in VGPRs ---
    v16bf a[16];
    const float* rowp = annot + (size_t)(bt0 + m) * HH;
    #pragma unroll
    for (int kc = 0; kc < 16; ++kc) {
        const float4* p0 = (const float4*)(rowp + kc * 32 + khalf * 16);
        const float4* p1 = (const float4*)(rowp + kc * 32 + khalf * 16 + 8);
        a[kc] = cvt16(p0[0], p0[1], p1[0], p1[1]);
    }
    // --- A operand for proc_loc (16x32 bf16) ---
    const __bf16* lrow = locA + (size_t)(bt0 + m) * LOCD;
    v8bf l0 = *(const v8bf*)(lrow + khalf * 16);
    v8bf l1 = *(const v8bf*)(lrow + khalf * 16 + 8);
    v16bf la;
    #pragma unroll
    for (int i = 0; i < 8; ++i) { la[i] = l0[i]; la[8 + i] = l1[i]; }

    float rs[8];
    #pragma unroll
    for (int j = 0; j < 8; ++j) rs[j] = 0.f;

    const v8f czero = {};

#ifdef HAVE_TDM
    const unsigned int lds0 = (unsigned int)(uintptr_t)(void*)&stage[0][0];
    const unsigned int lds1 = (unsigned int)(uintptr_t)(void*)&stage[1][0];
    const unsigned long long gbase = (unsigned long long)(uintptr_t)packW;

    if (wave == 0) {                                   // prologue: chunk 0
        tdm_load_chunk(lds0, gbase);
        __builtin_amdgcn_s_wait_tensorcnt(0);
    }
    __syncthreads();

    for (int nc = 0; nc < 32; ++nc) {
        const unsigned char* sb = stage[nc & 1];
        if (wave == 0 && nc + 1 < 32)                  // overlap next DMA
            tdm_load_chunk((nc & 1) ? lds0 : lds1,
                           gbase + (unsigned long long)(nc + 1) * WCHUNK_BYTES);

        v16bf lb = *(const v16bf*)(sb + lane * 32);
        v8f acc = __builtin_amdgcn_wmma_f32_16x16x32_bf16(
            false, la, false, lb, (short)0, czero, false, false);
        #pragma unroll
        for (int kc = 0; kc < 16; ++kc) {
            v16bf bb = *(const v16bf*)(sb + 1024 + kc * 1024 + lane * 32);
            acc = __builtin_amdgcn_wmma_f32_16x16x32_bf16(
                false, a[kc], false, bb, (short)0, acc, false, false);
        }
        int   h = nc * 16 + (lane & 15);
        float q = qcomb[b * HH + h];
        float v = vw[h];
        #pragma unroll
        for (int j = 0; j < 8; ++j)
            rs[j] += tanhf(acc[j] + q) * v;

        if (wave == 0) __builtin_amdgcn_s_wait_tensorcnt(0);
        __syncthreads();
    }
#else
    // Fallback: cooperative synchronous staging (no TDM builtin available).
    for (int nc = 0; nc < 32; ++nc) {
        const uint4* src = (const uint4*)(packW + (size_t)nc * WCHUNK_ELEMS);
        uint4* dst = (uint4*)&stage[0][0];
        for (int i = threadIdx.x; i < WCHUNK_BYTES / 16; i += 256)
            dst[i] = src[i];
        __syncthreads();
        const unsigned char* sb = stage[0];
        v16bf lb = *(const v16bf*)(sb + lane * 32);
        v8f acc = __builtin_amdgcn_wmma_f32_16x16x32_bf16(
            false, la, false, lb, (short)0, czero, false, false);
        #pragma unroll
        for (int kc = 0; kc < 16; ++kc) {
            v16bf bb = *(const v16bf*)(sb + 1024 + kc * 1024 + lane * 32);
            acc = __builtin_amdgcn_wmma_f32_16x16x32_bf16(
                false, a[kc], false, bb, (short)0, acc, false, false);
        }
        int   h = nc * 16 + (lane & 15);
        float q = qcomb[b * HH + h];
        float v = vw[h];
        #pragma unroll
        for (int j = 0; j < 8; ++j)
            rs[j] += tanhf(acc[j] + q) * v;
        __syncthreads();
    }
#endif

    // reduce across the 16 lanes sharing each row
    #pragma unroll
    for (int j = 0; j < 8; ++j) {
        float s = rs[j];
        s += __shfl_xor(s, 1);
        s += __shfl_xor(s, 2);
        s += __shfl_xor(s, 4);
        s += __shfl_xor(s, 8);
        rs[j] = s;
    }
    if ((lane & 15) == 0) {
        int mbase = khalf * 8;
        #pragma unroll
        for (int j = 0; j < 8; ++j)
            energy[bt0 + mbase + j] = rs[j];
    }
}

// ---------------------------------------------------------------------------
// Kernel 6: softmax over T (one block per batch row).
// ---------------------------------------------------------------------------
__global__ void softmax_kernel(const float* __restrict__ e,
                               float* __restrict__ out) {
    __shared__ float red[TT];
    int b = blockIdx.x, t = threadIdx.x;        // 512
    float v = e[b * TT + t];
    red[t] = v;
    __syncthreads();
    for (int s = 256; s > 0; s >>= 1) {
        if (t < s) red[t] = fmaxf(red[t], red[t + s]);
        __syncthreads();
    }
    float mx = red[0];
    __syncthreads();
    float ex = expf(v - mx);
    red[t] = ex;
    __syncthreads();
    for (int s = 256; s > 0; s >>= 1) {
        if (t < s) red[t] += red[t + s];
        __syncthreads();
    }
    out[b * TT + t] = ex / red[0];
}

// ---------------------------------------------------------------------------
// Kernel 7: new_context[b,d] = sum_t align[b,t] * annotations[b,t,d]
// ---------------------------------------------------------------------------
__global__ void context_kernel(const float* __restrict__ align,
                               const float* __restrict__ annot,
                               float* __restrict__ out) {
    __shared__ float al[TT];
    int b = blockIdx.x, t = threadIdx.x;        // 256
    al[t] = align[b * TT + t];
    al[t + 256] = align[b * TT + t + 256];
    __syncthreads();
    for (int d = t; d < HH; d += 256) {
        const float* base = annot + (size_t)b * TT * HH + d;
        float s = 0.f;
        for (int k = 0; k < TT; ++k)
            s += al[k] * base[(size_t)k * HH];
        out[b * HH + d] = s;
    }
}

// ---------------------------------------------------------------------------
// Launcher
// ---------------------------------------------------------------------------
extern "C" void kernel_launch(void* const* d_in, const int* in_sizes, int n_in,
                              void* d_out, int out_size, void* d_ws, size_t ws_size,
                              hipStream_t stream) {
    const float* memory    = (const float*)d_in[0];
    const float* context   = (const float*)d_in[1];
    const float* rnn_state = (const float*)d_in[2];
    const float* annot     = (const float*)d_in[3];
    const float* attvec    = (const float*)d_in[4];
    const float* W_ih      = (const float*)d_in[5];
    const float* W_hh      = (const float*)d_in[6];
    const float* b_ih      = (const float*)d_in[7];
    const float* b_hh      = (const float*)d_in[8];
    const float* loc_conv_w= (const float*)d_in[9];
    const float* loc_lin_w = (const float*)d_in[10];
    const float* loc_lin_b = (const float*)d_in[11];
    const float* query_w   = (const float*)d_in[12];
    const float* query_b   = (const float*)d_in[13];
    const float* annot_w   = (const float*)d_in[14];
    const float* annot_b   = (const float*)d_in[15];
    const float* v_w       = (const float*)d_in[16];

    float* out_rnn   = (float*)d_out;           // (B,H)
    float* out_ctx   = out_rnn + BB * HH;       // (B,H)
    float* out_align = out_ctx + BB * HH;       // (B,T)

    // workspace layout
    char* ws = (char*)d_ws;
    float*  qcomb    = (float*)(ws);                        // 128 KB
    float*  energies = (float*)(ws + (128 << 10));          // 128 KB
    __bf16* packW    = (__bf16*)(ws + (256 << 10));         // 544 KB
    __bf16* locbf    = (__bf16*)(ws + (1 << 20));           // 2 MB

    pack_w_kernel<<<(32 * WCHUNK_ELEMS) / 256, 256, 0, stream>>>(
        annot_w, loc_lin_w, packW);
    loc_conv_kernel<<<(BB * TT * LOCD) / 256, 256, 0, stream>>>(
        attvec, loc_conv_w, locbf);
    gru_kernel<<<BB, HH, 0, stream>>>(memory, context, rnn_state,
                                      W_ih, W_hh, b_ih, b_hh, out_rnn);
    qcomb_kernel<<<BB, HH, 0, stream>>>(out_rnn, query_w, query_b,
                                        annot_b, loc_lin_b, qcomb);
    energy_kernel<<<(BB * TT / 16) / 8, 256, 0, stream>>>(
        annot, locbf, packW, qcomb, v_w, energies);
    softmax_kernel<<<BB, TT, 0, stream>>>(energies, out_align);
    context_kernel<<<BB, 256, 0, stream>>>(out_align, annot, out_ctx);
}
